// DiffPool_25383256720185
// MI455X (gfx1250) — matrix operand order repaired and checked by reading, not compiled, not run on hardware
//
#include <hip/hip_runtime.h>
#include <hip/hip_bf16.h>
#include <math.h>

// ---------------- CDNA5 WMMA types ----------------
typedef _Float16 f16;
typedef __attribute__((ext_vector_type(16))) _Float16 v16h;
typedef __attribute__((ext_vector_type(8)))  _Float16 v8h;
typedef __attribute__((ext_vector_type(8)))  float    v8f;

// Problem constants (from reference)
constexpr int NB  = 16;    // batch
constexpr int NN  = 1024;  // nodes
constexpr int FIN = 64;
constexpr int HID = 128;
constexpr int OUTC= 8;
constexpr int KCL = 5;     // clusters
constexpr int KPAD= 32;    // cluster dim padded to a full 32-col wave tile
constexpr float BN_EPS = 1e-5f;

// =====================================================================
// Batched WMMA GEMM, all-f16 operands:
//   C[z] (MxN, f32) = A[z] (MxK f16 row-major) * Bt[z]^T (Bt is NxK f16,
//   pre-transposed so each lane's fragment is contiguous 16B runs)
// One wave computes a 16x32 output tile (two 16x16 WMMA tiles sharing the
// A fragment); 4 waves per block. Requirements (all launches satisfy):
//   M % 16 == 0, Ncols % 32 == 0, Kdim % 32 == 0.
// Software-pipelined: fragments for step k+1 are loaded before the WMMAs
// of step k so global_load_b128 overlaps the matrix pipe.
// =====================================================================
__global__ __launch_bounds__(128)
void gemm_wmma_h(const f16* __restrict__ A, const f16* __restrict__ Bt,
                 const float* __restrict__ bias, float* __restrict__ Cm,
                 int Ncols, int Kdim, int lda, int ldbt, int ldc,
                 long aO, long aI, long bO, long bI, long cO, long cI,
                 int Hdiv, int biasInner, int relu)
{
  const int lane = threadIdx.x & 31;
  const int wave = threadIdx.x >> 5;
  const int j0 = (blockIdx.x * 4 + wave) * 32;
  if (j0 >= Ncols) return;                       // wave-uniform exit
  const int i0 = blockIdx.y * 16;
  const int z  = blockIdx.z;
  const int outer = z / Hdiv, inner = z % Hdiv;
  const f16* Ab = A  + (size_t)outer * aO + (size_t)inner * aI;
  const f16* Bb = Bt + (size_t)outer * bO + (size_t)inner * bI;
  float*     Cb = Cm + (size_t)outer * cO + (size_t)inner * cI;

  const int l15   = lane & 15;
  const int koffA = (lane >> 4) * 8;
  const int koffB = (lane >> 4) * 8;
  const f16* Ap  = Ab + (size_t)(i0 + l15) * lda;
  const f16* Bp0 = Bb + (size_t)(j0 + l15) * ldbt;
  const f16* Bp1 = Bb + (size_t)(j0 + 16 + l15) * ldbt;

  auto ldA = [&](int kb) -> v16h {
    v8h lo = *(const v8h*)(Ap + kb + koffA);
    v8h hi = *(const v8h*)(Ap + kb + 16 + koffA);
    return __builtin_shufflevector(lo, hi, 0,1,2,3,4,5,6,7,8,9,10,11,12,13,14,15);
  };
  auto ldB = [&](const f16* p, int kb) -> v16h {
    v8h lo = *(const v8h*)(p + kb + koffB);
    v8h hi = *(const v8h*)(p + kb + 16 + koffB);
    return __builtin_shufflevector(lo, hi, 0,1,2,3,4,5,6,7,8,9,10,11,12,13,14,15);
  };

  v8f acc0 = {}, acc1 = {};
  v16h a = ldA(0), b = ldB(Bp0, 0), c = ldB(Bp1, 0);
  for (int kb = 32; kb < Kdim; kb += 32) {
    v16h an = ldA(kb), bn = ldB(Bp0, kb), cn = ldB(Bp1, kb);
    acc0 = __builtin_amdgcn_wmma_f32_16x16x32_f16(
        false, a, false, b, (short)0, acc0, false, false);
    acc1 = __builtin_amdgcn_wmma_f32_16x16x32_f16(
        false, a, false, c, (short)0, acc1, false, false);
    a = an; b = bn; c = cn;
  }
  acc0 = __builtin_amdgcn_wmma_f32_16x16x32_f16(
      false, a, false, b, (short)0, acc0, false, false);
  acc1 = __builtin_amdgcn_wmma_f32_16x16x32_f16(
      false, a, false, c, (short)0, acc1, false, false);

  const int rbase = i0 + ((lane >> 4) * 8);
  {
    const int col = j0 + l15;
    const float bv = bias ? bias[(size_t)inner * biasInner + col] : 0.f;
#pragma unroll
    for (int r = 0; r < 8; ++r) {
      float v = acc0[r] + bv;
      if (relu) v = fmaxf(v, 0.f);
      Cb[(size_t)(rbase + r) * ldc + col] = v;
    }
  }
  {
    const int col = j0 + 16 + l15;
    const float bv = bias ? bias[(size_t)inner * biasInner + col] : 0.f;
#pragma unroll
    for (int r = 0; r < 8; ++r) {
      float v = acc1[r] + bv;
      if (relu) v = fmaxf(v, 0.f);
      Cb[(size_t)(rbase + r) * ldc + col] = v;
    }
  }
}

// ---------------- f32 -> f16 linear convert ----------------------------
__global__ void to_f16(const float* __restrict__ in, f16* __restrict__ out, long total)
{
  long idx = (long)blockIdx.x * blockDim.x + threadIdx.x;
  if (idx < total) out[idx] = (f16)in[idx];
}

// ---- weight transpose + zero-pad: out[colsOut][rows] = in[rows][cols]^T
__global__ void transpose_pad_f16(const float* __restrict__ in, f16* __restrict__ out,
                                  int rows, int cols, int colsOut)
{
  int idx = blockIdx.x * blockDim.x + threadIdx.x;
  int total = colsOut * rows;
  if (idx >= total) return;
  int r = idx % rows, c = idx / rows;
  out[idx] = (c < cols) ? (f16)in[(size_t)r * cols + c] : (f16)0.f;
}

// ---- per-(batch,head) transpose to f16: [B][rows][Hh*C] -> [B][Hh][C][rows]
__global__ void split_transpose_f16(const float* __restrict__ in, f16* __restrict__ out,
                                    int rows, int Hh, int C)
{
  long idx = (long)blockIdx.x * blockDim.x + threadIdx.x;
  long total = (long)NB * Hh * C * rows;
  if (idx >= total) return;
  int r = (int)(idx % rows); long t = idx / rows;
  int c = (int)(t % C); t /= C;
  int h = (int)(t % Hh); int b = (int)(t / Hh);
  out[idx] = (f16)in[((size_t)b * rows + r) * ((size_t)Hh * C) + (size_t)h * C + c];
}

__global__ void pad_bias(const float* __restrict__ in, float* __restrict__ out,
                         int n, int nOut)
{
  int i = threadIdx.x;
  if (i < nOut) out[i] = (i < n) ? in[i] : 0.f;
}

// ---------------- a_src / a_dst: per-(b,n,h) dot over C -----------------
__global__ void attdot(const float* __restrict__ Hf, const float* __restrict__ att,
                       float* __restrict__ outp, int rows, int Hh, int C)
{
  int idx = blockIdx.x * blockDim.x + threadIdx.x;
  int total = NB * rows * Hh;
  if (idx >= total) return;
  int h = idx % Hh; int n = (idx / Hh) % rows; int b = idx / (Hh * rows);
  const float* hp = Hf + ((size_t)(b * rows + n) * Hh + h) * C;
  const float* ap = att + (size_t)h * C;
  float s = 0.f;
  for (int c = 0; c < C; ++c) s += hp[c] * ap[c];
  outp[idx] = s;
}

// -------- attention rows: leaky + mask(adj|diag) + softmax -> P (f16) ---
__global__ __launch_bounds__(256)
void attn_rows(const float* __restrict__ adj, const float* __restrict__ asrc,
               const float* __restrict__ adst, f16* __restrict__ P,
               int Nnodes, int Hh)
{
  __shared__ float red[256];
  const int i = blockIdx.x, bh = blockIdx.y;
  const int b = bh / Hh, h = bh % Hh;
  const int tid = threadIdx.x;
  const float* arow = adj + ((size_t)b * Nnodes + i) * Nnodes;
  const float ad = adst[((size_t)b * Nnodes + i) * Hh + h];

  float mx = -3.4e38f;
  for (int j = tid; j < Nnodes; j += 256) {
    if (arow[j] != 0.f || j == i) {
      float v = asrc[((size_t)b * Nnodes + j) * Hh + h] + ad;
      v = v > 0.f ? v : 0.2f * v;
      mx = fmaxf(mx, v);
    }
  }
  red[tid] = mx; __syncthreads();
  for (int s = 128; s > 0; s >>= 1) { if (tid < s) red[tid] = fmaxf(red[tid], red[tid + s]); __syncthreads(); }
  mx = red[0]; __syncthreads();

  float sum = 0.f;
  for (int j = tid; j < Nnodes; j += 256) {
    if (arow[j] != 0.f || j == i) {
      float v = asrc[((size_t)b * Nnodes + j) * Hh + h] + ad;
      v = v > 0.f ? v : 0.2f * v;
      sum += __expf(v - mx);
    }
  }
  red[tid] = sum; __syncthreads();
  for (int s = 128; s > 0; s >>= 1) { if (tid < s) red[tid] += red[tid + s]; __syncthreads(); }
  sum = red[0]; __syncthreads();
  const float inv = 1.f / sum;

  f16* prow = P + ((size_t)bh * Nnodes + i) * Nnodes;   // [b][h][i][j]
  for (int j = tid; j < Nnodes; j += 256) {
    float p = 0.f;
    if (arow[j] != 0.f || j == i) {
      float v = asrc[((size_t)b * Nnodes + j) * Hh + h] + ad;
      v = v > 0.f ? v : 0.2f * v;
      p = __expf(v - mx) * inv;
    }
    prow[j] = (f16)p;
  }
}

// ---------------- BatchNorm1d (eval, stats over axis-1) + opt relu ------
// in has row stride ldin (>= C); out is dense C.
__global__ void bn_act(const float* __restrict__ in, float* __restrict__ outp,
                       const float* __restrict__ g, const float* __restrict__ bb,
                       const float* __restrict__ mn, const float* __restrict__ vr,
                       int nodes, int C, int ldin, int relu)
{
  long idx = (long)blockIdx.x * blockDim.x + threadIdx.x;
  long total = (long)NB * nodes * C;
  if (idx >= total) return;
  int c = (int)(idx % C);
  int node = (int)((idx / C) % nodes);
  long bn = idx / ((long)C * nodes);
  float x = in[((size_t)bn * nodes + node) * ldin + c];
  float y = (x - mn[node]) * rsqrtf(vr[node] + BN_EPS) * g[node] + bb[node];
  if (relu) y = fmaxf(y, 0.f);
  outp[idx] = y;
}

// ---------------- GCN degree normalization ------------------------------
__global__ __launch_bounds__(256)
void degree_inv(const float* __restrict__ adj, float* __restrict__ dinv, int Nnodes)
{
  __shared__ float red[256];
  const int b = blockIdx.y, i = blockIdx.x, tid = threadIdx.x;
  const float* row = adj + ((size_t)b * Nnodes + i) * Nnodes;
  float s = 0.f;
  for (int j = tid; j < Nnodes; j += 256) { float a = row[j]; if (j == i) a = 1.f; s += a; }
  red[tid] = s; __syncthreads();
  for (int st = 128; st > 0; st >>= 1) { if (tid < st) red[tid] += red[tid + st]; __syncthreads(); }
  if (tid == 0) dinv[(size_t)b * Nnodes + i] = rsqrtf(fmaxf(red[0], 1.f));
}

__global__ void norm_adj(const float* __restrict__ adj, const float* __restrict__ dinv,
                         f16* __restrict__ adjn, int Nnodes)
{
  long idx = (long)blockIdx.x * blockDim.x + threadIdx.x;
  long total = (long)NB * Nnodes * Nnodes;
  if (idx >= total) return;
  int j = (int)(idx % Nnodes); long t = idx / Nnodes;
  int i = (int)(t % Nnodes); int b = (int)(t / Nnodes);
  float a = adj[idx]; if (i == j) a = 1.f;
  adjn[idx] = (f16)(a * dinv[(size_t)b * Nnodes + i] * dinv[(size_t)b * Nnodes + j]);
}

// -------- softmax over K=5, twice (PyG re-softmax), + entropy loss ------
__global__ void softmax5_ent(const float* __restrict__ in, float* __restrict__ s_out,
                             float* __restrict__ s2_out, float* __restrict__ ent_acc)
{
  int idx = blockIdx.x * blockDim.x + threadIdx.x;
  int total = NB * NN;
  if (idx >= total) return;
  const float* x = in + (size_t)idx * KCL;
  float v[KCL], mx = -3.4e38f;
#pragma unroll
  for (int k = 0; k < KCL; ++k) { v[k] = x[k]; mx = fmaxf(mx, v[k]); }
  float sum = 0.f;
#pragma unroll
  for (int k = 0; k < KCL; ++k) { v[k] = __expf(v[k] - mx); sum += v[k]; }
  float inv = 1.f / sum;
#pragma unroll
  for (int k = 0; k < KCL; ++k) { v[k] *= inv; s_out[(size_t)idx * KCL + k] = v[k]; }
  // second softmax (dense_diff_pool internal)
  mx = -3.4e38f;
#pragma unroll
  for (int k = 0; k < KCL; ++k) mx = fmaxf(mx, v[k]);
  sum = 0.f;
#pragma unroll
  for (int k = 0; k < KCL; ++k) { v[k] = __expf(v[k] - mx); sum += v[k]; }
  inv = 1.f / sum;
  float e = 0.f;
#pragma unroll
  for (int k = 0; k < KCL; ++k) {
    v[k] *= inv;
    s2_out[(size_t)idx * KCL + k] = v[k];
    e -= v[k] * logf(v[k] + 1e-15f);
  }
  atomicAdd(ent_acc, e / (float)(NB * NN));
}

// ---------------- t = adj @ s2  (B,N,5) --------------------------------
__global__ __launch_bounds__(256)
void adj_s(const float* __restrict__ adj, const float* __restrict__ s2,
           float* __restrict__ tb, int Nnodes)
{
  __shared__ float red[256];
  const int b = blockIdx.y, i = blockIdx.x, tid = threadIdx.x;
  const float* arow = adj + ((size_t)b * Nnodes + i) * Nnodes;
  float acc[KCL] = {0.f, 0.f, 0.f, 0.f, 0.f};
  for (int j = tid; j < Nnodes; j += 256) {
    float a = arow[j];
    if (a != 0.f) {
      const float* sj = s2 + ((size_t)b * Nnodes + j) * KCL;
#pragma unroll
      for (int k = 0; k < KCL; ++k) acc[k] += a * sj[k];
    }
  }
  for (int k = 0; k < KCL; ++k) {
    red[tid] = acc[k]; __syncthreads();
    for (int st = 128; st > 0; st >>= 1) { if (tid < st) red[tid] += red[tid + st]; __syncthreads(); }
    if (tid == 0) tb[((size_t)b * Nnodes + i) * KCL + k] = red[0];
    __syncthreads();
  }
}

// ---------------- x_c[b,k,c] = sum_n s2[b,n,k] z_local[b,n,c] ----------
__global__ __launch_bounds__(128)
void pool_x(const float* __restrict__ s2, const float* __restrict__ zl,
            float* __restrict__ xc, int Nnodes, int C)
{
  const int b = blockIdx.y, k = blockIdx.x, c = threadIdx.x;
  float a = 0.f;
  for (int n = 0; n < Nnodes; ++n)
    a += s2[((size_t)b * Nnodes + n) * KCL + k] * zl[((size_t)b * Nnodes + n) * C + c];
  xc[((size_t)b * KCL + k) * C + c] = a;
}

// ---------------- adj_c[b,k,l] = sum_n s2[b,n,k] t[b,n,l] --------------
__global__ void pool_adj(const float* __restrict__ s2, const float* __restrict__ tb,
                         float* __restrict__ adjc, int Nnodes)
{
  const int b = blockIdx.x; const int t = threadIdx.x;
  if (t >= KCL * KCL) return;
  const int k = t / KCL, l = t % KCL;
  float a = 0.f;
  for (int n = 0; n < Nnodes; ++n)
    a += s2[((size_t)b * Nnodes + n) * KCL + k] * tb[((size_t)b * Nnodes + n) * KCL + l];
  adjc[(size_t)b * KCL * KCL + t] = a;
}

// ---------------- link loss: sum (adj - s2 s2^T)^2 ---------------------
__global__ __launch_bounds__(256)
void link_loss_k(const float* __restrict__ adj, const float* __restrict__ s2,
                 float* __restrict__ acc, int Nnodes)
{
  __shared__ float red[256];
  const int b = blockIdx.y, i = blockIdx.x, tid = threadIdx.x;
  float si[KCL];
#pragma unroll
  for (int k = 0; k < KCL; ++k) si[k] = s2[((size_t)b * Nnodes + i) * KCL + k];
  const float* arow = adj + ((size_t)b * Nnodes + i) * Nnodes;
  float part = 0.f;
  for (int j = tid; j < Nnodes; j += 256) {
    const float* sj = s2 + ((size_t)b * Nnodes + j) * KCL;
    float d = 0.f;
#pragma unroll
    for (int k = 0; k < KCL; ++k) d += si[k] * sj[k];
    float v = arow[j] - d;
    part += v * v;
  }
  red[tid] = part; __syncthreads();
  for (int st = 128; st > 0; st >>= 1) { if (tid < st) red[tid] += red[tid + st]; __syncthreads(); }
  if (tid == 0) atomicAdd(acc, red[0]);
}

// ---------------- small dense GAT on K=5 pooled nodes ------------------
__global__ __launch_bounds__(128)
void gat_small(const float* __restrict__ feat, const float* __restrict__ adjc,
               const float* __restrict__ asrc, const float* __restrict__ adst,
               const float* __restrict__ bias, float* __restrict__ outp,
               int rows, int Hh, int C, int relu)
{
  const int i = blockIdx.x, bh = blockIdx.y;
  const int b = bh / Hh, h = bh % Hh;
  const int c = threadIdx.x;
  const float ad = adst[((size_t)(b * rows + i)) * Hh + h];
  float al[8];
  float mx = -3.4e38f;
  for (int j = 0; j < rows; ++j) {
    float a = adjc[((size_t)b * rows + i) * rows + j];
    bool m = (a != 0.f) || (j == i);
    float v = asrc[((size_t)(b * rows + j)) * Hh + h] + ad;
    v = v > 0.f ? v : 0.2f * v;
    al[j] = m ? v : -3.4e38f;
    mx = fmaxf(mx, al[j]);
  }
  float sum = 0.f;
  for (int j = 0; j < rows; ++j) { al[j] = __expf(al[j] - mx); sum += al[j]; }
  const float inv = 1.f / sum;
  float acc = 0.f;
  for (int j = 0; j < rows; ++j)
    acc += al[j] * inv * feat[((size_t)(b * rows + j)) * ((size_t)Hh * C) + (size_t)h * C + c];
  acc += bias[(size_t)h * C + c];
  if (relu) acc = fmaxf(acc, 0.f);
  outp[((size_t)(b * rows + i)) * ((size_t)Hh * C) + (size_t)h * C + c] = acc;
}

// ---------------- tiny matmul: Y[b,r,:] = X[b,r,:] @ W (+bias)(+relu) --
__global__ void small_mm(const float* __restrict__ X, const float* __restrict__ W,
                         const float* __restrict__ bias, float* __restrict__ Y,
                         int rows, int Kd, int Ncols, int relu)
{
  int idx = blockIdx.x * blockDim.x + threadIdx.x;
  int total = NB * rows * Ncols;
  if (idx >= total) return;
  int o = idx % Ncols; int r = (idx / Ncols) % rows; int b = idx / (Ncols * rows);
  const float* xp = X + (size_t)(b * rows + r) * Kd;
  float a = bias ? bias[o] : 0.f;
  for (int k = 0; k < Kd; ++k) a += xp[k] * W[(size_t)k * Ncols + o];
  if (relu) a = fmaxf(a, 0.f);
  Y[idx] = a;
}

// ---------------- mean over nodes --------------------------------------
__global__ void mean_nodes(const float* __restrict__ X, float* __restrict__ Y,
                           int rows, int C)
{
  int idx = blockIdx.x * blockDim.x + threadIdx.x;
  if (idx >= NB * C) return;
  int b = idx / C, c = idx % C;
  float a = 0.f;
  for (int r = 0; r < rows; ++r) a += X[(size_t)(b * rows + r) * C + c];
  Y[idx] = a / (float)rows;
}

__global__ void zero2(float* p) { p[0] = 0.f; p[1] = 0.f; }

__global__ void finalize_losses(const float* __restrict__ accs,
                                float* __restrict__ o_link, float* __restrict__ o_ent)
{
  o_link[0] = sqrtf(accs[0]) / (float)((size_t)NB * NN * NN);
  o_ent[0]  = accs[1];
}

// =====================================================================
extern "C" void kernel_launch(void* const* d_in, const int* in_sizes, int n_in,
                              void* d_out, int out_size, void* d_ws, size_t ws_size,
                              hipStream_t stream)
{
  // Input indices (jax sorted-key flattening of setup_inputs dict):
  const float* ADJ = (const float*)d_in[0];
  const float* BN1C_B=(const float*)d_in[1],  *BN1C_G=(const float*)d_in[2],  *BN1C_M=(const float*)d_in[3],  *BN1C_V=(const float*)d_in[4];
  const float* BN1E_B=(const float*)d_in[5],  *BN1E_G=(const float*)d_in[6],  *BN1E_M=(const float*)d_in[7],  *BN1E_V=(const float*)d_in[8];
  const float* BN1P_B=(const float*)d_in[9],  *BN1P_G=(const float*)d_in[10], *BN1P_M=(const float*)d_in[11], *BN1P_V=(const float*)d_in[12];
  const float* BN2C_B=(const float*)d_in[13], *BN2C_G=(const float*)d_in[14], *BN2C_M=(const float*)d_in[15], *BN2C_V=(const float*)d_in[16];
  const float* BN2E_B=(const float*)d_in[17], *BN2E_G=(const float*)d_in[18], *BN2E_M=(const float*)d_in[19], *BN2E_V=(const float*)d_in[20];
  const float* BN2P_B=(const float*)d_in[21], *BN2P_G=(const float*)d_in[22], *BN2P_M=(const float*)d_in[23], *BN2P_V=(const float*)d_in[24];
  const float* C1W=(const float*)d_in[25], *C1AD=(const float*)d_in[26], *C1AS=(const float*)d_in[27], *C1B=(const float*)d_in[28];
  const float* C2W=(const float*)d_in[29], *C2AD=(const float*)d_in[30], *C2AS=(const float*)d_in[31], *C2B=(const float*)d_in[32];
  const float* FC1W=(const float*)d_in[33], *FC1B=(const float*)d_in[34];
  const float* FC2W=(const float*)d_in[35], *FC2B=(const float*)d_in[36];
  const float* G1W=(const float*)d_in[37], *G1AD=(const float*)d_in[38], *G1AS=(const float*)d_in[39], *G1B=(const float*)d_in[40];
  const float* G2W=(const float*)d_in[41], *G2AD=(const float*)d_in[42], *G2AS=(const float*)d_in[43], *G2B=(const float*)d_in[44];
  const float* P1W=(const float*)d_in[45], *P1B=(const float*)d_in[46];
  const float* P2W=(const float*)d_in[47], *P2B=(const float*)d_in[48];
  const float* X  = (const float*)d_in[49];

  // Workspace allocator
  char* wsp = (char*)d_ws;
  size_t cur = 0;
  auto alloc = [&](size_t bytes) -> char* {
    char* p = wsp + cur;
    cur += (bytes + 255) & ~(size_t)255;
    return p;
  };
  // f16 staging
  f16*   xh   = (f16*)  alloc((size_t)NB*NN*FIN*2);
  f16*   g1Wt = (f16*)  alloc((size_t)256*FIN*2);
  f16*   g2Wt = (f16*)  alloc((size_t)128*256*2);
  f16*   p1Wt = (f16*)  alloc((size_t)128*FIN*2);
  f16*   p2Wt = (f16*)  alloc((size_t)KPAD*128*2);
  f16*   h1t  = (f16*)  alloc((size_t)NB*2*128*NN*2);
  f16*   h2t  = (f16*)  alloc((size_t)NB*128*NN*2);
  f16*   xw1t = (f16*)  alloc((size_t)NB*128*NN*2);
  f16*   sw2t = (f16*)  alloc((size_t)NB*KPAD*NN*2);
  f16*   z1rh = (f16*)  alloc((size_t)NB*NN*256*2);
  f16*   s1rh = (f16*)  alloc((size_t)NB*NN*128*2);
  f16*   P1   = (f16*)  alloc((size_t)NB*2*NN*NN*2);
  f16*   P2   = (f16*)  alloc((size_t)NB*NN*NN*2);
  f16*   adjn = (f16*)  alloc((size_t)NB*NN*NN*2);
  // f32 intermediates
  float* h1   = (float*)alloc((size_t)NB*NN*256*4);
  float* as1  = (float*)alloc((size_t)NB*NN*2*4);
  float* ad1  = (float*)alloc((size_t)NB*NN*2*4);
  float* z1   = (float*)alloc((size_t)NB*NN*256*4);
  float* z1r  = (float*)alloc((size_t)NB*NN*256*4);
  float* h2   = (float*)alloc((size_t)NB*NN*128*4);
  float* as2  = (float*)alloc((size_t)NB*NN*4);
  float* ad2  = (float*)alloc((size_t)NB*NN*4);
  float* zlp  = (float*)alloc((size_t)NB*NN*128*4);
  float* dinv = (float*)alloc((size_t)NB*NN*4);
  float* xw1  = (float*)alloc((size_t)NB*NN*128*4);
  float* sp1  = (float*)alloc((size_t)NB*NN*128*4);
  float* s1r  = (float*)alloc((size_t)NB*NN*128*4);
  float* sw2p = (float*)alloc((size_t)NB*NN*KPAD*4);
  float* sp2p = (float*)alloc((size_t)NB*NN*KPAD*4);
  float* sp2b = (float*)alloc((size_t)NB*NN*KCL*4);
  float* pb2p = (float*)alloc((size_t)KPAD*4);
  float* s2b  = (float*)alloc((size_t)NB*NN*KCL*4);
  float* tb   = (float*)alloc((size_t)NB*NN*KCL*4);
  float* xc   = (float*)alloc((size_t)NB*KCL*128*4);
  float* adjc = (float*)alloc((size_t)NB*KCL*KCL*4);
  float* accs = (float*)alloc(256);
  float* m1h  = (float*)alloc((size_t)NB*KCL*256*4);
  float* asc1 = (float*)alloc((size_t)NB*KCL*2*4);
  float* adc1 = (float*)alloc((size_t)NB*KCL*2*4);
  float* m1   = (float*)alloc((size_t)NB*KCL*256*4);
  float* m1r  = (float*)alloc((size_t)NB*KCL*256*4);
  float* m2h  = (float*)alloc((size_t)NB*KCL*128*4);
  float* asc2 = (float*)alloc((size_t)NB*KCL*4);
  float* adc2 = (float*)alloc((size_t)NB*KCL*4);
  float* m2   = (float*)alloc((size_t)NB*KCL*128*4);
  float* zlm  = (float*)alloc((size_t)NB*128*4);
  float* zmm  = (float*)alloc((size_t)NB*128*4);
  float* omb  = (float*)alloc((size_t)NB*128*4);

  // Output layout (tuple flattened in return order)
  float* outp = (float*)d_out;
  float* o_zl = outp;                       // (16,1024,128)
  float* o_zm = o_zl + (size_t)NB*NN*128;   // (16,5,128)
  float* o_s  = o_zm + (size_t)NB*KCL*128;  // (16,1024,5)
  float* o_ol = o_s  + (size_t)NB*NN*KCL;   // (16,8)
  float* o_om = o_ol + (size_t)NB*OUTC;     // (16,8)
  float* o_lk = o_om + (size_t)NB*OUTC;     // scalar
  float* o_en = o_lk + 1;                   // scalar

  zero2<<<1, 1, 0, stream>>>(accs);

  // ---- operand prep ----
  to_f16<<<((long)NB*NN*FIN + 255)/256, 256, 0, stream>>>(X, xh, (long)NB*NN*FIN);
  transpose_pad_f16<<<(256*FIN + 255)/256, 256, 0, stream>>>(G1W, g1Wt, FIN, 256, 256);
  transpose_pad_f16<<<(128*256 + 255)/256, 256, 0, stream>>>(G2W, g2Wt, 256, 128, 128);
  transpose_pad_f16<<<(128*FIN + 255)/256, 256, 0, stream>>>(P1W, p1Wt, FIN, 128, 128);
  transpose_pad_f16<<<(KPAD*128 + 255)/256, 256, 0, stream>>>(P2W, p2Wt, 128, KCL, KPAD);
  pad_bias<<<1, 32, 0, stream>>>(P2B, pb2p, KCL, KPAD);

  // ===== local embed branch =====
  // h1 = x @ g1_W   (1024x64 @ 64x256)
  gemm_wmma_h<<<dim3(2, NN/16, NB), 128, 0, stream>>>(
      xh, g1Wt, nullptr, h1, 256, FIN, FIN, FIN, 256,
      (long)NN*FIN, 0, 0, 0, (long)NN*256, 0, 1, 0, 0);
  attdot<<<(NB*NN*2 + 255)/256, 256, 0, stream>>>(h1, G1AS, as1, NN, 2, HID);
  attdot<<<(NB*NN*2 + 255)/256, 256, 0, stream>>>(h1, G1AD, ad1, NN, 2, HID);
  attn_rows<<<dim3(NN, NB*2), 256, 0, stream>>>(ADJ, as1, ad1, P1, NN, 2);
  split_transpose_f16<<<((long)NB*2*128*NN + 255)/256, 256, 0, stream>>>(h1, h1t, NN, 2, 128);
  // z1 = P1 @ h1 + g1_b    (per (b,h): 1024x1024 @ 1024x128)
  gemm_wmma_h<<<dim3(1, NN/16, NB*2), 128, 0, stream>>>(
      P1, h1t, G1B, z1, 128, NN, NN, NN, 256,
      (long)2*NN*NN, (long)NN*NN, (long)2*128*NN, (long)128*NN,
      (long)NN*256, 128, 2, 128, 0);
  bn_act<<<((long)NB*NN*256 + 255)/256, 256, 0, stream>>>(
      z1, z1r, BN1E_G, BN1E_B, BN1E_M, BN1E_V, NN, 256, 256, 1);
  to_f16<<<((long)NB*NN*256 + 255)/256, 256, 0, stream>>>(z1r, z1rh, (long)NB*NN*256);
  // h2 = z1r @ g2_W   (1024x256 @ 256x128)
  gemm_wmma_h<<<dim3(1, NN/16, NB), 128, 0, stream>>>(
      z1rh, g2Wt, nullptr, h2, 128, 256, 256, 256, 128,
      (long)NN*256, 0, 0, 0, (long)NN*128, 0, 1, 0, 0);
  attdot<<<(NB*NN + 255)/256, 256, 0, stream>>>(h2, G2AS, as2, NN, 1, HID);
  attdot<<<(NB*NN + 255)/256, 256, 0, stream>>>(h2, G2AD, ad2, NN, 1, HID);
  attn_rows<<<dim3(NN, NB), 256, 0, stream>>>(ADJ, as2, ad2, P2, NN, 1);
  split_transpose_f16<<<((long)NB*128*NN + 255)/256, 256, 0, stream>>>(h2, h2t, NN, 1, 128);
  // zlp = relu(P2 @ h2 + g2_b)
  gemm_wmma_h<<<dim3(1, NN/16, NB), 128, 0, stream>>>(
      P2, h2t, G2B, zlp, 128, NN, NN, NN, 128,
      (long)NN*NN, 0, (long)128*NN, 0, (long)NN*128, 0, 1, 0, 1);
  bn_act<<<((long)NB*NN*128 + 255)/256, 256, 0, stream>>>(
      zlp, o_zl, BN2E_G, BN2E_B, BN2E_M, BN2E_V, NN, 128, 128, 1);

  // ===== pooling assignment branch (GCN) =====
  degree_inv<<<dim3(NN, NB), 256, 0, stream>>>(ADJ, dinv, NN);
  norm_adj<<<((long)NB*NN*NN + 255)/256, 256, 0, stream>>>(ADJ, dinv, adjn, NN);
  gemm_wmma_h<<<dim3(1, NN/16, NB), 128, 0, stream>>>(
      xh, p1Wt, nullptr, xw1, 128, FIN, FIN, FIN, 128,
      (long)NN*FIN, 0, 0, 0, (long)NN*128, 0, 1, 0, 0);
  split_transpose_f16<<<((long)NB*128*NN + 255)/256, 256, 0, stream>>>(xw1, xw1t, NN, 1, 128);
  gemm_wmma_h<<<dim3(1, NN/16, NB), 128, 0, stream>>>(
      adjn, xw1t, P1B, sp1, 128, NN, NN, NN, 128,
      (long)NN*NN, 0, (long)128*NN, 0, (long)NN*128, 0, 1, 0, 0);
  bn_act<<<((long)NB*NN*128 + 255)/256, 256, 0, stream>>>(
      sp1, s1r, BN1P_G, BN1P_B, BN1P_M, BN1P_V, NN, 128, 128, 1);
  to_f16<<<((long)NB*NN*128 + 255)/256, 256, 0, stream>>>(s1r, s1rh, (long)NB*NN*128);
  // sw2p = s1r @ p2_W  (padded to 32 cols; cols 5..31 are exact zeros)
  gemm_wmma_h<<<dim3(1, NN/16, NB), 128, 0, stream>>>(
      s1rh, p2Wt, nullptr, sw2p, KPAD, 128, 128, 128, KPAD,
      (long)NN*128, 0, 0, 0, (long)NN*KPAD, 0, 1, 0, 0);
  split_transpose_f16<<<((long)NB*KPAD*NN + 255)/256, 256, 0, stream>>>(sw2p, sw2t, NN, 1, KPAD);
  gemm_wmma_h<<<dim3(1, NN/16, NB), 128, 0, stream>>>(
      adjn, sw2t, pb2p, sp2p, KPAD, NN, NN, NN, KPAD,
      (long)NN*NN, 0, (long)KPAD*NN, 0, (long)NN*KPAD, 0, 1, 0, 0);
  bn_act<<<((long)NB*NN*KCL + 255)/256, 256, 0, stream>>>(
      sp2p, sp2b, BN2P_G, BN2P_B, BN2P_M, BN2P_V, NN, KCL, KPAD, 0);
  softmax5_ent<<<(NB*NN + 255)/256, 256, 0, stream>>>(sp2b, o_s, s2b, accs + 1);

  // ===== diff pool =====
  adj_s<<<dim3(NN, NB), 256, 0, stream>>>(ADJ, s2b, tb, NN);
  pool_x<<<dim3(KCL, NB), 128, 0, stream>>>(s2b, o_zl, xc, NN, 128);
  pool_adj<<<NB, 32, 0, stream>>>(s2b, tb, adjc, NN);
  link_loss_k<<<dim3(NN, NB), 256, 0, stream>>>(ADJ, s2b, accs, NN);

  // ===== coarse branch (K=5 nodes) =====
  small_mm<<<(NB*KCL*256 + 255)/256, 256, 0, stream>>>(xc, C1W, nullptr, m1h, KCL, 128, 256, 0);
  attdot<<<(NB*KCL*2 + 255)/256, 256, 0, stream>>>(m1h, C1AS, asc1, KCL, 2, HID);
  attdot<<<(NB*KCL*2 + 255)/256, 256, 0, stream>>>(m1h, C1AD, adc1, KCL, 2, HID);
  gat_small<<<dim3(KCL, NB*2), 128, 0, stream>>>(m1h, adjc, asc1, adc1, C1B, m1, KCL, 2, HID, 0);
  bn_act<<<((long)NB*KCL*256 + 255)/256, 256, 0, stream>>>(
      m1, m1r, BN1C_G, BN1C_B, BN1C_M, BN1C_V, KCL, 256, 256, 1);
  small_mm<<<(NB*KCL*128 + 255)/256, 256, 0, stream>>>(m1r, C2W, nullptr, m2h, KCL, 256, 128, 0);
  attdot<<<(NB*KCL + 255)/256, 256, 0, stream>>>(m2h, C2AS, asc2, KCL, 1, HID);
  attdot<<<(NB*KCL + 255)/256, 256, 0, stream>>>(m2h, C2AD, adc2, KCL, 1, HID);
  gat_small<<<dim3(KCL, NB), 128, 0, stream>>>(m2h, adjc, asc2, adc2, C2B, m2, KCL, 1, HID, 0);
  bn_act<<<((long)NB*KCL*128 + 255)/256, 256, 0, stream>>>(
      m2, o_zm, BN2C_G, BN2C_B, BN2C_M, BN2C_V, KCL, 128, 128, 1);

  // ===== readout heads =====
  mean_nodes<<<(NB*128 + 255)/256, 256, 0, stream>>>(o_zl, zlm, NN, 128);
  mean_nodes<<<(NB*128 + 255)/256, 256, 0, stream>>>(o_zm, zmm, KCL, 128);
  small_mm<<<(NB*128 + 255)/256, 256, 0, stream>>>(zmm, FC1W, FC1B, omb, 1, 128, 128, 1);
  small_mm<<<(NB*OUTC + 255)/256, 256, 0, stream>>>(omb, FC2W, FC2B, o_om, 1, 128, OUTC, 0);
  small_mm<<<(NB*OUTC + 255)/256, 256, 0, stream>>>(zlm, FC2W, FC2B, o_ol, 1, 128, OUTC, 0);
  finalize_losses<<<1, 1, 0, stream>>>(accs, o_lk, o_en);
}
